// MultiheadAttention_695784702387
// MI455X (gfx1250) — compile-verified, matrix operand order checked by reading, not run on hardware
//
#include <hip/hip_runtime.h>
#include <hip/hip_bf16.h>
#include <cstdint>

typedef __attribute__((ext_vector_type(16))) __bf16 v16bf;
typedef __attribute__((ext_vector_type(8)))  float  v8f;
typedef __attribute__((ext_vector_type(4)))  unsigned int u32x4;
typedef __attribute__((ext_vector_type(8)))  int      i32x8;
typedef __attribute__((ext_vector_type(4)))  int      i32x4;

#define B_ 4
#define T_ 2048
#define E_ 1024
#define H_ 16
#define D_ 64

#if __has_builtin(__builtin_amdgcn_tensor_load_to_lds) && \
    __has_builtin(__builtin_amdgcn_s_wait_tensorcnt)
#define HAVE_TDM 1
#else
#define HAVE_TDM 0
#endif

// ---------- WMMA helpers (gfx1250, wave32) ----------

__device__ __forceinline__ v8f wmma_bf16(v16bf a, v16bf b, v8f c) {
  // 8 args: (neg_a, A, neg_b, B, c_mod, C, reuse_a, reuse_b)
  return __builtin_amdgcn_wmma_f32_16x16x32_bf16(false, a, false, b, (short)0, c,
                                                 false, false);
}

// A-fragment: 16x32 bf16, row r = lane%16 contiguous; K split per ISA 7.12.2.
__device__ __forceinline__ v16bf frag_row_bf16(const __bf16* __restrict__ base,
                                               int stride) {
  const int lane = threadIdx.x & 31;
  const int r = lane & 15, hf = lane >> 4;
  const __bf16* p = base + r * stride;
  v16bf f;
#pragma unroll
  for (int i = 0; i < 8; ++i) f[i] = p[hf * 8 + i];
#pragma unroll
  for (int i = 0; i < 8; ++i) f[8 + i] = p[16 + hf * 8 + i];
  return f;
}

// Same pattern, f32 source (softmax probs in LDS), convert to bf16 inline.
__device__ __forceinline__ v16bf frag_row_f32cvt(const float* __restrict__ base,
                                                 int stride) {
  const int lane = threadIdx.x & 31;
  const int r = lane & 15, hf = lane >> 4;
  const float* p = base + r * stride;
  v16bf f;
#pragma unroll
  for (int i = 0; i < 8; ++i) f[i] = (__bf16)p[hf * 8 + i];
#pragma unroll
  for (int i = 0; i < 8; ++i) f[8 + i] = (__bf16)p[16 + hf * 8 + i];
  return f;
}

// B-fragment from K-major (row = k) storage: element (k, n) = base[k*stride + n].
__device__ __forceinline__ v16bf frag_col_bf16(const __bf16* __restrict__ base,
                                               int stride) {
  const int lane = threadIdx.x & 31;
  const int n = lane & 15, hf = lane >> 4;
  v16bf f;
#pragma unroll
  for (int i = 0; i < 8; ++i) f[i] = base[(hf * 8 + i) * stride + n];
#pragma unroll
  for (int i = 0; i < 8; ++i) f[8 + i] = base[(16 + hf * 8 + i) * stride + n];
  return f;
}

#if HAVE_TDM
// Tensor Data Mover: DMA a 64x64 bf16 tile (row stride 64 elems in memory) into
// LDS at lds_off, padding each 32-DWORD row with 4 DWORDs -> LDS row stride 72
// bf16 elems. D# layout per CDNA5 ISA ch.8 (group0 128b, group1 256b).
// This toolchain's builtin takes 6 args (g0, g1, g2, g3, g4, cpol).
__device__ __forceinline__ void tdm_load_tile64x64(const __bf16* gsrc,
                                                   unsigned int lds_off) {
  const unsigned long long ga = (unsigned long long)(uintptr_t)gsrc;
  u32x4 g0;
  g0[0] = 1u;                                   // count=1, user descriptor
  g0[1] = lds_off;                              // lds_addr (bytes)
  g0[2] = (unsigned int)ga;                     // global_addr[31:0]
  g0[3] = (unsigned int)((ga >> 32) & 0x1FFFFFFull) | (2u << 30);  // [56:32]|type=2
  i32x8 g1;
  g1[0] = (1 << 16)       // data_size = 2 bytes
        | (1 << 20)       // pad_enable
        | (4 << 22)       // pad_interval: 32 DWORDs (one 64-elem bf16 row)
        | (3 << 25);      // pad_amount: 4 DWORDs  -> LDS row stride 72 elems
  g1[1] = (64 << 16);     // tensor_dim0 = 64
  g1[2] = (64 << 16);     // tensor_dim1 = 64 (rows available from tile start)
  g1[3] = (64 << 16);     // tile_dim0 = 64
  g1[4] = 64;             // tile_dim1 = 64, tile_dim2 = 0
  g1[5] = 64;             // tensor_dim0_stride = 64 elems
  g1[6] = 0;
  g1[7] = 0;
  const i32x4 z4 = {0, 0, 0, 0};
  const i32x8 z8 = {0, 0, 0, 0, 0, 0, 0, 0};
  __builtin_amdgcn_tensor_load_to_lds(g0, g1, z4, z4, z8, 0);
}
#endif

// ---------- Kernel 1: QKV projection  Y = X @ W^T  (f32 in, bf16 [B,H,T,D] out)
// 128x128 macro-tile, k-step 32, double-buffered LDS so global->LDS (with f32->
// bf16 convert) overlaps the WMMA pipeline. 8 waves, each owns a 32x64 tile.

__global__ __launch_bounds__(256) void qkv_proj_kernel(
    const float* __restrict__ Xq, const float* __restrict__ Xk,
    const float* __restrict__ Xv, const float* __restrict__ Wq,
    const float* __restrict__ Wk, const float* __restrict__ Wv,
    __bf16* __restrict__ Qb, __bf16* __restrict__ Kb, __bf16* __restrict__ Vb) {
  const int z = blockIdx.z;
  const float* X = (z == 0) ? Xq : ((z == 1) ? Xk : Xv);
  const float* W = (z == 0) ? Wq : ((z == 1) ? Wk : Wv);
  __bf16* O = (z == 0) ? Qb : ((z == 1) ? Kb : Vb);

  const int n0 = blockIdx.x * 128;
  const int m0 = blockIdx.y * 128;

  __shared__ __align__(16) __bf16 sA[2][128][40];
  __shared__ __align__(16) __bf16 sB[2][128][40];

  const int tid = threadIdx.x;
  const int wave = tid >> 5;
  const int wm = wave >> 1;  // 4 wave-groups over M (32 rows each)
  const int wn = wave & 1;   // 2 wave-groups over N (64 cols each)

  v8f acc[2][4] = {};

  auto load_tiles = [&](int buf, int k0) {
#pragma unroll
    for (int it = 0; it < 4; ++it) {
      int i = tid + it * 256;
      int e = i * 4;
      int row = e >> 5, col = e & 31;
      float4 v = *(const float4*)(X + (size_t)(m0 + row) * E_ + k0 + col);
      __bf16* d = &sA[buf][row][col];
      d[0] = (__bf16)v.x; d[1] = (__bf16)v.y; d[2] = (__bf16)v.z; d[3] = (__bf16)v.w;
      float4 w = *(const float4*)(W + (size_t)(n0 + row) * E_ + k0 + col);
      __bf16* dw = &sB[buf][row][col];
      dw[0] = (__bf16)w.x; dw[1] = (__bf16)w.y; dw[2] = (__bf16)w.z; dw[3] = (__bf16)w.w;
    }
  };

  load_tiles(0, 0);
  int cur = 0;
  for (int kt = 0; kt < E_ / 32; ++kt) {
    __syncthreads();  // tile 'cur' ready
    if (kt + 1 < E_ / 32) load_tiles(cur ^ 1, (kt + 1) * 32);
    v16bf af[2], bfr[4];
#pragma unroll
    for (int i = 0; i < 2; ++i)
      af[i] = frag_row_bf16(&sA[cur][wm * 32 + i * 16][0], 40);
#pragma unroll
    for (int j = 0; j < 4; ++j)
      bfr[j] = frag_row_bf16(&sB[cur][wn * 64 + j * 16][0], 40);
#pragma unroll
    for (int i = 0; i < 2; ++i)
#pragma unroll
      for (int j = 0; j < 4; ++j) acc[i][j] = wmma_bf16(af[i], bfr[j], acc[i][j]);
    cur ^= 1;
  }

  const int lane = tid & 31;
  const int cn = lane & 15, chf = lane >> 4;
#pragma unroll
  for (int i = 0; i < 2; ++i)
#pragma unroll
    for (int j = 0; j < 4; ++j)
#pragma unroll
      for (int r = 0; r < 8; ++r) {
        int m = m0 + wm * 32 + i * 16 + chf * 8 + r;  // global row = b*T + t
        int n = n0 + wn * 64 + j * 16 + cn;           // global col = h*D + d
        int bb = m >> 11, t = m & (T_ - 1);
        int hh = n >> 6, d = n & (D_ - 1);
        O[(((size_t)bb * H_ + hh) * T_ + t) * D_ + d] = (__bf16)acc[i][j][r];
      }
}

// ---------- Kernel 2: attention. One block = (b, 16 query rows), loops heads.
// ~291 KB CDNA5 LDS: exact scores + block-local head-mean + TDM-staged V tiles.

__global__ __launch_bounds__(256) void attn_kernel(
    const __bf16* __restrict__ Qb, const __bf16* __restrict__ Kb,
    const __bf16* __restrict__ Vb, __bf16* __restrict__ AV,
    float* __restrict__ Pmean) {
  extern __shared__ char smem[];
  float* scoresF = (float*)smem;                              // 16*2048 f32 128KB
  float* psum    = (float*)(smem + 16 * T_ * 4);              // 16*2048 f32 128KB
  __bf16* qtile  = (__bf16*)(smem + 2 * 16 * T_ * 4);         // 16*64 bf16   2KB
  float* pv      = (float*)(smem + 2 * 16 * T_ * 4 + 16 * D_ * 2);  // 8*256 f32 8KB
  __bf16* vtile  = (__bf16*)(smem + 2 * 16 * T_ * 4 + 16 * D_ * 2 + 8 * 256 * 4);
  // vtile: 2 buffers x 64 rows x stride 72 bf16 = 18 KB (double-buffered DMA)

  const int b = blockIdx.y;
  const int t0 = blockIdx.x * 16;
  const int tid = threadIdx.x;
  const int wave = tid >> 5;
  const int lane = tid & 31;
  const int cn = lane & 15, chf = lane >> 4;

  for (int i = tid; i < 16 * T_; i += 256) psum[i] = 0.0f;

  for (int h = 0; h < H_; ++h) {
    const size_t headoff = ((size_t)b * H_ + h) * T_ * D_;
    const __bf16* Qh = Qb + headoff;
    const __bf16* Kh = Kb + headoff;
    const __bf16* Vh = Vb + headoff;

    // stage Q tile (16 x 64) into LDS
    for (int i = tid; i < (16 * D_) / 8; i += 256) {
      int e = i * 8;
      int r = e >> 6, c = e & 63;
      *(float4*)&qtile[r * D_ + c] =
          *(const float4*)(Qh + (size_t)(t0 + r) * D_ + c);
    }
    __syncthreads();

    // scores = (Q K^T) / sqrt(D); 128 s-tiles of 16, round-robin over 8 waves
    for (int st = wave; st < T_ / 16; st += 8) {
      if (st + 8 < T_ / 16)
        __builtin_prefetch((const void*)(Kh + (size_t)(st + 8) * 16 * D_), 0, 1);
      v8f acc = {};
#pragma unroll
      for (int k0 = 0; k0 < D_; k0 += 32) {
        v16bf a = frag_row_bf16(qtile + k0, D_);
        v16bf bk = frag_row_bf16(Kh + (size_t)st * 16 * D_ + k0, D_);  // B = K^T
        acc = wmma_bf16(a, bk, acc);
      }
#pragma unroll
      for (int r = 0; r < 8; ++r)
        scoresF[(chf * 8 + r) * T_ + st * 16 + cn] = acc[r] * 0.125f;
    }
    __syncthreads();

    // softmax: 16 threads per row, width-16 shuffle reductions (wave32-safe)
    {
      const int row = tid >> 4;
      const int c0 = tid & 15;
      float* srow = scoresF + row * T_;
      float mx = -3.0e38f;
      for (int c = c0; c < T_; c += 16) mx = fmaxf(mx, srow[c]);
#pragma unroll
      for (int off = 8; off > 0; off >>= 1) mx = fmaxf(mx, __shfl_xor(mx, off, 16));
      float sum = 0.0f;
      for (int c = c0; c < T_; c += 16) {
        float e = __expf(srow[c] - mx);
        srow[c] = e;
        sum += e;
      }
#pragma unroll
      for (int off = 8; off > 0; off >>= 1) sum += __shfl_xor(sum, off, 16);
      const float inv = 1.0f / sum;
      float* prow = psum + row * T_;
      for (int c = c0; c < T_; c += 16) {
        float p = srow[c] * inv;
        srow[c] = p;      // final prob, consumed by PV WMMA
        prow[c] += p;     // head-mean accumulation (block-local, no atomics)
      }
    }
    __syncthreads();

    // attn_vec = P @ V. V is staged 64x64 per step: TDM DMA double-buffered so
    // the tensor engine overlaps the WMMAs. 8 waves: d-subtile (w&3) x s-half.
    {
      const int dsub = wave & 3;
      const int shalf = wave >> 2;
      v8f acc = {};
      int cur = 0;
#if HAVE_TDM
      if (wave == 0) tdm_load_tile64x64(Vh, (unsigned int)(uintptr_t)vtile);
#else
      for (int i = tid; i < 512; i += 256) {
        int e = i * 8;
        int r = e >> 6, c = e & 63;
        *(float4*)&vtile[r * 72 + c] = *(const float4*)(Vh + (size_t)r * D_ + c);
      }
#endif
      for (int it = 0; it < T_ / 64; ++it) {
#if HAVE_TDM
        if (wave == 0) __builtin_amdgcn_s_wait_tensorcnt(0);
#endif
        __syncthreads();  // vtile[cur] ready for everyone
        if (it + 1 < T_ / 64) {
          __bf16* nxt = vtile + (cur ^ 1) * (64 * 72);
          const __bf16* gsrc = Vh + (size_t)(it + 1) * 64 * D_;
#if HAVE_TDM
          if (wave == 0) tdm_load_tile64x64(gsrc, (unsigned int)(uintptr_t)nxt);
#else
          for (int i = tid; i < 512; i += 256) {
            int e = i * 8;
            int r = e >> 6, c = e & 63;
            *(float4*)&nxt[r * 72 + c] = *(const float4*)(gsrc + (size_t)r * D_ + c);
          }
#endif
        }
        const int sl = shalf * 32;                    // this wave's 32-row slice
        v16bf a = frag_row_f32cvt(scoresF + it * 64 + sl, T_);
        v16bf bv = frag_col_bf16(vtile + cur * (64 * 72) + sl * 72 + dsub * 16, 72);
        acc = wmma_bf16(a, bv, acc);
        __syncthreads();  // everyone done with vtile[cur] before it is rewritten
        cur ^= 1;
      }
      float* pvw = pv + wave * 256;  // wave = shalf*4 + dsub
#pragma unroll
      for (int r = 0; r < 8; ++r) pvw[(chf * 8 + r) * 16 + cn] = acc[r];
    }
    __syncthreads();

    // reduce the two s-halves and emit AV in [B, T, H*D] bf16
    for (int i = tid; i < 16 * D_; i += 256) {
      int r = i >> 6, d = i & 63;
      int dsub = d >> 4, dn = d & 15;
      float v0 = pv[(0 * 4 + dsub) * 256 + r * 16 + dn];
      float v1 = pv[(1 * 4 + dsub) * 256 + r * 16 + dn];
      AV[((size_t)b * T_ + t0 + r) * (H_ * D_) + (size_t)h * D_ + d] =
          (__bf16)(v0 + v1);
    }
    __syncthreads();
  }

  // mean over heads -> [B, T, S] f32
  const float invH = 1.0f / (float)H_;
  for (int i = tid; i < 16 * T_; i += 256) {
    int r = i >> 11, c = i & (T_ - 1);
    Pmean[((size_t)b * T_ + t0 + r) * T_ + c] = psum[i] * invH;
  }
}

// ---------- Kernel 3: output projection  x = AV @ Wo^T (bf16 x f32 -> f32),
// double-buffered like kernel 1.

__global__ __launch_bounds__(256) void out_proj_kernel(
    const __bf16* __restrict__ A, const float* __restrict__ Wo,
    float* __restrict__ Y) {
  const int n0 = blockIdx.x * 128;
  const int m0 = blockIdx.y * 128;
  __shared__ __align__(16) __bf16 sA[2][128][40];
  __shared__ __align__(16) __bf16 sB[2][128][40];
  const int tid = threadIdx.x;
  const int wave = tid >> 5;
  const int wm = wave >> 1, wn = wave & 1;

  v8f acc[2][4] = {};

  auto load_tiles = [&](int buf, int k0) {
#pragma unroll
    for (int it = 0; it < 2; ++it) {  // A already bf16: 8 elems / 16B per op
      int i = tid + it * 256;
      int e = i * 8;
      int row = e >> 5, col = e & 31;
      *(float4*)&sA[buf][row][col] =
          *(const float4*)(A + (size_t)(m0 + row) * E_ + k0 + col);
    }
#pragma unroll
    for (int it = 0; it < 4; ++it) {  // Wo f32 -> bf16
      int i = tid + it * 256;
      int e = i * 4;
      int row = e >> 5, col = e & 31;
      float4 w = *(const float4*)(Wo + (size_t)(n0 + row) * E_ + k0 + col);
      __bf16* dw = &sB[buf][row][col];
      dw[0] = (__bf16)w.x; dw[1] = (__bf16)w.y; dw[2] = (__bf16)w.z; dw[3] = (__bf16)w.w;
    }
  };

  load_tiles(0, 0);
  int cur = 0;
  for (int kt = 0; kt < E_ / 32; ++kt) {
    __syncthreads();
    if (kt + 1 < E_ / 32) load_tiles(cur ^ 1, (kt + 1) * 32);
    v16bf af[2], bfr[4];
#pragma unroll
    for (int i = 0; i < 2; ++i)
      af[i] = frag_row_bf16(&sA[cur][wm * 32 + i * 16][0], 40);
#pragma unroll
    for (int j = 0; j < 4; ++j)
      bfr[j] = frag_row_bf16(&sB[cur][wn * 64 + j * 16][0], 40);
#pragma unroll
    for (int i = 0; i < 2; ++i)
#pragma unroll
      for (int j = 0; j < 4; ++j) acc[i][j] = wmma_bf16(af[i], bfr[j], acc[i][j]);
    cur ^= 1;
  }

  const int lane = tid & 31;
  const int cn = lane & 15, chf = lane >> 4;
#pragma unroll
  for (int i = 0; i < 2; ++i)
#pragma unroll
    for (int j = 0; j < 4; ++j)
#pragma unroll
      for (int r = 0; r < 8; ++r) {
        int m = m0 + wm * 32 + i * 16 + chf * 8 + r;
        int n = n0 + wn * 64 + j * 16 + cn;
        Y[(size_t)m * E_ + n] = acc[i][j][r];
      }
}

// ---------- host ----------

extern "C" void kernel_launch(void* const* d_in, const int* in_sizes, int n_in,
                              void* d_out, int out_size, void* d_ws, size_t ws_size,
                              hipStream_t stream) {
  const float* query = (const float*)d_in[0];
  const float* key_  = (const float*)d_in[1];
  const float* value = (const float*)d_in[2];
  const float* Wq = (const float*)d_in[3];
  const float* Wk = (const float*)d_in[4];
  const float* Wv = (const float*)d_in[5];
  const float* Wo = (const float*)d_in[6];

  float* x_out = (float*)d_out;                      // [B,T,E] f32
  float* p_out = x_out + (size_t)B_ * T_ * E_;       // [B,T,T] f32

  __bf16* Qb = (__bf16*)d_ws;
  const size_t headElems = (size_t)B_ * H_ * T_ * D_;  // 8,388,608
  __bf16* Kb = Qb + headElems;
  __bf16* Vb = Kb + headElems;
  __bf16* AV = Vb + headElems;  // [B,T,H*D] bf16 — total ws use: 64 MB

  // 1) QKV projections (z selects q/k/v)
  qkv_proj_kernel<<<dim3(E_ / 128, (B_ * T_) / 128, 3), 256, 0, stream>>>(
      query, key_, value, Wq, Wk, Wv, Qb, Kb, Vb);

  // 2) attention with head loop + head-mean probs (needs >64KB dynamic LDS)
  const int attnLds = (2 * 16 * T_ * 4) + (16 * D_ * 2) + (8 * 256 * 4)
                    + (2 * 64 * 72 * 2);  // 290816 B
  (void)hipFuncSetAttribute((const void*)attn_kernel,
                            hipFuncAttributeMaxDynamicSharedMemorySize, attnLds);
  attn_kernel<<<dim3(T_ / 16, B_), 256, attnLds, stream>>>(Qb, Kb, Vb, AV, p_out);

  // 3) output projection
  out_proj_kernel<<<dim3(E_ / 128, (B_ * T_) / 128), 256, 0, stream>>>(AV, Wo, x_out);
}
